// DKBATNet_82222853914934
// MI455X (gfx1250) — matrix operand order repaired and checked by reading, not compiled, not run on hardware
//
#include <hip/hip_runtime.h>
#include <hip/hip_bf16.h>

// ---------------------------------------------------------------------------
// DKBAT 2-level relational GAT for MI455X (gfx1250, wave32, WMMA).
// Factorized attention: per-node projections via bf16 WMMA GEMMs (operands
// pre-converted to bf16 once), then memory-bound edge gather/softmax/scatter
// passes with L2-resident f32 atomics.
// ---------------------------------------------------------------------------

typedef __attribute__((ext_vector_type(16))) __bf16 v16bf;
typedef __attribute__((ext_vector_type(8)))  __bf16 v8bf;
typedef __attribute__((ext_vector_type(4)))  __bf16 v4bf;
typedef __attribute__((ext_vector_type(8)))  float  v8f;

__device__ __forceinline__ __bf16 f2bf(float f) {
    unsigned u = __builtin_bit_cast(unsigned, f);
    unsigned r = u + 0x7FFFu + ((u >> 16) & 1u);           // round-to-nearest-even
    return __builtin_bit_cast(__bf16, (unsigned short)(r >> 16));
}

// ---------------------------------------------------------------------------
// WMMA GEMM: C[M,NC] = A[M,K](bf16) * Wbf[:,off:off+K].T (bf16) (+f32 bias).
// One wave -> 16x64 C strip (4 col tiles, A-fragment reuse). Inner loop is
// pure global_load_b128 -> v_wmma_f32_16x16x32_bf16 (no conversion ALU).
// ---------------------------------------------------------------------------
template <int K, int NC, bool BIAS>
__global__ void wmma_gemm(const __bf16* __restrict__ A, int M,
                          const __bf16* __restrict__ W, int ldw,
                          const float* __restrict__ bias,
                          float* __restrict__ C, int totalWaves)
{
    int wid = blockIdx.x * (blockDim.x >> 5) + (threadIdx.x >> 5);
    if (wid >= totalWaves) return;
    constexpr int WPR = NC / 64;                 // waves per 16-row strip
    int rowTile = wid / WPR;
    int colBase = (wid % WPR) * 64;
    int lane = threadIdx.x & 31;
    int half = lane >> 4;                        // 0 | 1
    int l16  = lane & 15;

    int mA = rowTile * 16 + l16;
    if (mA > M - 1) mA = M - 1;                  // clamp loads, guard stores
    const __bf16* arow = A + (size_t)mA * K;
    const int kbA = half * 16;                   // A frag: 16 contiguous K per half
    const int kbB = half * 16;                   // B frag: 16 contiguous K per half

    v8f acc[4];
#pragma unroll
    for (int i = 0; i < 4; ++i)
#pragma unroll
        for (int j = 0; j < 8; ++j) acc[i][j] = 0.f;

    for (int k0 = 0; k0 < K; k0 += 32) {
        __builtin_prefetch(arow + k0 + 128, 0, 3);
        v8bf lo = *(const v8bf*)(arow + k0 + kbA);
        v8bf hi = *(const v8bf*)(arow + k0 + kbA + 8);
        v16bf a = __builtin_shufflevector(lo, hi, 0,1,2,3,4,5,6,7,
                                                  8,9,10,11,12,13,14,15);
#pragma unroll
        for (int ct = 0; ct < 4; ++ct) {
            v16bf b = *(const v16bf*)(W + (size_t)(colBase + ct * 16 + l16) * ldw
                                        + k0 + kbB);
            acc[ct] = __builtin_amdgcn_wmma_f32_16x16x32_bf16(
                false, a, false, b, (short)0, acc[ct], false, false);
        }
    }

#pragma unroll
    for (int ct = 0; ct < 4; ++ct) {
        int n = colBase + ct * 16 + l16;
        float bv = BIAS ? bias[n] : 0.f;
#pragma unroll
        for (int i = 0; i < 8; ++i) {
            int mm = rowTile * 16 + half * 8 + i;   // C/D layout: M = i + 8*half
            if (mm < M) C[(size_t)mm * NC + n] = acc[ct][i] + bv;
        }
    }
}

// ---------------------------------------------------------------------------
// Helpers
// ---------------------------------------------------------------------------
__device__ __forceinline__ void load8(float* c, const float* __restrict__ p) {
    float4 a = *(const float4*)p, b = *(const float4*)(p + 4);
    c[0]=a.x; c[1]=a.y; c[2]=a.z; c[3]=a.w; c[4]=b.x; c[5]=b.y; c[6]=b.z; c[7]=b.w;
}
__device__ __forceinline__ void add8(float* c, const float* __restrict__ p) {
    float4 a = *(const float4*)p, b = *(const float4*)(p + 4);
    c[0]+=a.x; c[1]+=a.y; c[2]+=a.z; c[3]+=a.w; c[4]+=b.x; c[5]+=b.y; c[6]+=b.z; c[7]+=b.w;
}
__device__ __forceinline__ float half_reduce(float v) {   // 16-lane group sum
    v += __shfl_xor(v, 1); v += __shfl_xor(v, 2);
    v += __shfl_xor(v, 4); v += __shfl_xor(v, 8);
    return v;
}
__device__ __forceinline__ float full_reduce(float v) {   // 32-lane sum
    v = half_reduce(v); v += __shfl_xor(v, 16); return v;
}

__global__ void fzero(float* __restrict__ p, long n) {
    long i = (long)blockIdx.x * blockDim.x + threadIdx.x;
    if (i < n) p[i] = 0.f;
}

// f32 -> bf16 bulk conversion, 4 elements/thread
__global__ void f32_to_bf16_vec(const float* __restrict__ src,
                                __bf16* __restrict__ dst, long n4) {
    long i = (long)blockIdx.x * blockDim.x + threadIdx.x;
    if (i >= n4) return;
    float4 v = *(const float4*)(src + i * 4);
    v4bf o; o[0]=f2bf(v.x); o[1]=f2bf(v.y); o[2]=f2bf(v.z); o[3]=f2bf(v.w);
    *(v4bf*)(dst + i * 4) = o;
}

// g_aux (bf16): g rows 0..R-1 plus zero row R
__global__ void build_gaux_bf(const float* __restrict__ g,
                              __bf16* __restrict__ gaux, int R) {
    int i = blockIdx.x * blockDim.x + threadIdx.x;
    int tot = (R + 1) * 128;
    if (i < tot) gaux[i] = (i < R * 128) ? f2bf(g[i]) : f2bf(0.f);
}

// per-row L2 normalize, write bf16 (consumed only as GEMM A operand)
__global__ void l2norm_rows_bf(const float* __restrict__ X,
                               __bf16* __restrict__ Y, int Nn) {
    int wid = blockIdx.x * (blockDim.x >> 5) + (threadIdx.x >> 5);
    if (wid >= Nn) return;
    int lane = threadIdx.x & 31;
    float4 v = *(const float4*)(X + (size_t)wid * 128 + lane * 4);
    float ss = full_reduce(v.x*v.x + v.y*v.y + v.z*v.z + v.w*v.w);
    float sc = 1.f / fmaxf(sqrtf(ss), 1e-12f);
    v4bf o; o[0]=f2bf(v.x*sc); o[1]=f2bf(v.y*sc); o[2]=f2bf(v.z*sc); o[3]=f2bf(v.w*sc);
    *(v4bf*)(Y + (size_t)wid * 128 + lane * 4) = o;
}

// Pass A: c_e = bias + Arow[src] + Acol[dst] + G[k1] + G[k2];
// a = <att, c> per head; e = exp(leaky(a, .01)); denom[dst,h] += e.
__global__ void edge_pass_a(const int* __restrict__ src, const int* __restrict__ dst,
                            const int* __restrict__ k1,  const int* __restrict__ k2,
                            const float* __restrict__ Ar, const float* __restrict__ Ac,
                            const float* __restrict__ G,  const float* __restrict__ bias,
                            const float* __restrict__ att,
                            float* __restrict__ evals, float* __restrict__ denom, int E)
{
    int wid = blockIdx.x * (blockDim.x >> 5) + (threadIdx.x >> 5);
    if (wid >= E) return;
    int lane = threadIdx.x & 31, t0 = lane * 8;
    int s = src[wid], d = dst[wid], r1 = k1[wid], r2 = k2[wid];
    float c[8], av[8];
    load8(c, bias + t0);
    add8(c, Ar + (size_t)s * 256 + t0);
    add8(c, Ac + (size_t)d * 256 + t0);
    add8(c, G + (size_t)r1 * 256 + t0);
    add8(c, G + (size_t)r2 * 256 + t0);
    load8(av, att + t0);
    float part = 0.f;
#pragma unroll
    for (int j = 0; j < 8; ++j) part += c[j] * av[j];
    float a = half_reduce(part);                      // per-head (16-lane) dot
    float lk = a >= 0.f ? a : 0.01f * a;
    float e = expf(lk);
    if ((lane & 15) == 0) {
        int h = lane >> 4;
        evals[2 * (size_t)wid + h] = e;
        atomicAdd(&denom[2 * (size_t)d + h], e);
    }
}

// Pass B: recompute c_e, alpha = e/denom[dst], scatter alpha*c into acc[dst].
__global__ void edge_pass_b(const int* __restrict__ src, const int* __restrict__ dst,
                            const int* __restrict__ k1,  const int* __restrict__ k2,
                            const float* __restrict__ Ar, const float* __restrict__ Ac,
                            const float* __restrict__ G,  const float* __restrict__ bias,
                            const float* __restrict__ evals, const float* __restrict__ denom,
                            float* __restrict__ acc, int E)
{
    int wid = blockIdx.x * (blockDim.x >> 5) + (threadIdx.x >> 5);
    if (wid >= E) return;
    int lane = threadIdx.x & 31, t0 = lane * 8, h = lane >> 4;
    int s = src[wid], d = dst[wid], r1 = k1[wid], r2 = k2[wid];
    float c[8];
    load8(c, bias + t0);
    add8(c, Ar + (size_t)s * 256 + t0);
    add8(c, Ac + (size_t)d * 256 + t0);
    add8(c, G + (size_t)r1 * 256 + t0);
    add8(c, G + (size_t)r2 * 256 + t0);
    float alpha = evals[2 * (size_t)wid + h] / denom[2 * (size_t)d + h];
    float* ap = acc + (size_t)d * 256 + t0;
#pragma unroll
    for (int j = 0; j < 8; ++j) atomicAdd(&ap[j], alpha * c[j]);
}

// Gate: al = sigmoid(<aW[:128],h_in> + <aW[128:],h_out> + ab) per (n,h);
// h = al*h_in + (1-al)*h_out; leaky(.2); per-head L2 norm.
// BFOUT=true -> bf16 output (level-1, consumed as GEMM A); else f32.
template <bool BFOUT>
__global__ void gate_norm(const float* __restrict__ accIn, const float* __restrict__ accOut,
                          const float* __restrict__ aW, const float* __restrict__ ab,
                          void* __restrict__ Hout, int Nn)
{
    int wid = blockIdx.x * (blockDim.x >> 5) + (threadIdx.x >> 5);
    if (wid >= Nn) return;
    int lane = threadIdx.x & 31, t0 = lane * 8, o0 = t0 & 127;
    float hin[8], hout[8], wi[8], wo[8];
    load8(hin,  accIn  + (size_t)wid * 256 + t0);
    load8(hout, accOut + (size_t)wid * 256 + t0);
    load8(wi, aW + o0);
    load8(wo, aW + 128 + o0);
    float z = 0.f;
#pragma unroll
    for (int j = 0; j < 8; ++j) z += wi[j] * hin[j] + wo[j] * hout[j];
    z = half_reduce(z) + ab[0];
    float al = 1.f / (1.f + expf(-z));
    float y[8]; float ss = 0.f;
#pragma unroll
    for (int j = 0; j < 8; ++j) {
        float v = al * hin[j] + (1.f - al) * hout[j];
        v = v >= 0.f ? v : 0.2f * v;
        y[j] = v; ss += v * v;
    }
    ss = half_reduce(ss);
    float sc = 1.f / fmaxf(sqrtf(ss), 1e-12f);
    if (BFOUT) {
        v8bf o;
#pragma unroll
        for (int j = 0; j < 8; ++j) o[j] = f2bf(y[j] * sc);
        *(v8bf*)((__bf16*)Hout + (size_t)wid * 256 + t0) = o;
    } else {
        float* op = (float*)Hout + (size_t)wid * 256 + t0;
#pragma unroll
        for (int j = 0; j < 8; ++j) op[j] = y[j] * sc;
    }
}

// h_prime = mean over heads of l2norm(Pe[n,:] + h2[n,h,:])
__global__ void final_merge(const float* __restrict__ Pe, const float* __restrict__ H2,
                            float* __restrict__ out, int Nn)
{
    int wid = blockIdx.x * (blockDim.x >> 5) + (threadIdx.x >> 5);
    if (wid >= Nn) return;
    int lane = threadIdx.x & 31;
    float4 pe = *(const float4*)(Pe + (size_t)wid * 128 + lane * 4);
    float oacc[4] = {0.f, 0.f, 0.f, 0.f};
#pragma unroll
    for (int h = 0; h < 2; ++h) {
        float4 q = *(const float4*)(H2 + (size_t)wid * 256 + h * 128 + lane * 4);
        float v[4] = {pe.x + q.x, pe.y + q.y, pe.z + q.z, pe.w + q.w};
        float ss = full_reduce(v[0]*v[0] + v[1]*v[1] + v[2]*v[2] + v[3]*v[3]);
        float sc = 1.f / fmaxf(sqrtf(ss), 1e-12f);
#pragma unroll
        for (int j = 0; j < 4; ++j) oacc[j] += v[j] * sc;
    }
    float4 o = {oacc[0]*0.5f, oacc[1]*0.5f, oacc[2]*0.5f, oacc[3]*0.5f};
    *(float4*)(out + (size_t)wid * 128 + lane * 4) = o;
}

// ---------------------------------------------------------------------------
// Host side
// ---------------------------------------------------------------------------
static inline void* carveB(char*& base, size_t bytes) {
    void* p = (void*)base;
    base += ((bytes + 255) / 256) * 256;
    return p;
}

template <int K, int NC, bool BIAS>
static void launch_gemm(const __bf16* A, int M, const __bf16* W, int ldw,
                        const float* bias, float* C, hipStream_t s) {
    int tiles = (M + 15) / 16;
    int waves = tiles * (NC / 64);
    int blocks = (waves + 7) / 8;
    wmma_gemm<K, NC, BIAS><<<blocks, 256, 0, s>>>(A, M, W, ldw, bias, C, waves);
}

static void conv_bf16(const float* src, __bf16* dst, long n, hipStream_t s) {
    long n4 = n / 4;
    f32_to_bf16_vec<<<(int)((n4 + 255) / 256), 256, 0, s>>>(src, dst, n4);
}

static void att_dir(const __bf16* Hbf, int Dh,
                    const float* W, const float* b, const float* att,
                    const int* src, const int* dst, const int* k1, const int* k2,
                    const __bf16* gauxbf, __bf16* Wbf,
                    float* Arow, float* Acol, float* Gtab,
                    float* evals, float* denom, float* acc,
                    int N, int R, int E, hipStream_t s)
{
    int ldw = 2 * Dh + 128;
    conv_bf16(W, Wbf, 256L * ldw, s);            // stage this layer's weights once
    if (Dh == 128) {
        launch_gemm<128, 256, false>(Hbf, N, Wbf,       ldw, nullptr, Arow, s);
        launch_gemm<128, 256, false>(Hbf, N, Wbf + 128, ldw, nullptr, Acol, s);
    } else {
        launch_gemm<256, 256, false>(Hbf, N, Wbf,       ldw, nullptr, Arow, s);
        launch_gemm<256, 256, false>(Hbf, N, Wbf + 256, ldw, nullptr, Acol, s);
    }
    launch_gemm<128, 256, false>(gauxbf, R + 1, Wbf + 2 * Dh, ldw, nullptr, Gtab, s);

    long nd = 2L * N, na = 256L * N;
    fzero<<<(int)((nd + 255) / 256), 256, 0, s>>>(denom, nd);
    fzero<<<(int)((na + 255) / 256), 256, 0, s>>>(acc,   na);

    int eb = (E + 7) / 8;
    edge_pass_a<<<eb, 256, 0, s>>>(src, dst, k1, k2, Arow, Acol, Gtab, b, att,
                                   evals, denom, E);
    edge_pass_b<<<eb, 256, 0, s>>>(src, dst, k1, k2, Arow, Acol, Gtab, b,
                                   evals, denom, acc, E);
}

extern "C" void kernel_launch(void* const* d_in, const int* in_sizes, int n_in,
                              void* d_out, int out_size, void* d_ws, size_t ws_size,
                              hipStream_t stream)
{
    const float* x    = (const float*)d_in[0];
    const float* g    = (const float*)d_in[1];
    const int*   eidx = (const int*)d_in[2];
    const int*   etyp = (const int*)d_in[3];
    int N = in_sizes[0] / 128;
    int R = in_sizes[1] / 128;
    int E = in_sizes[2] / 2;
    const int *erow = eidx, *ecol = eidx + E;
    const int *ek1 = etyp,  *ek2 = etyp + E;

    const float *W_ii=(const float*)d_in[4],  *b_ii=(const float*)d_in[5],  *att_ii=(const float*)d_in[6];
    const float *W_io=(const float*)d_in[7],  *b_io=(const float*)d_in[8],  *att_io=(const float*)d_in[9];
    const float *W_oi=(const float*)d_in[10], *b_oi=(const float*)d_in[11], *att_oi=(const float*)d_in[12];
    const float *W_oo=(const float*)d_in[13], *b_oo=(const float*)d_in[14], *att_oo=(const float*)d_in[15];
    const float *aW_in=(const float*)d_in[16],  *ab_in=(const float*)d_in[17];
    const float *aW_out=(const float*)d_in[18], *ab_out=(const float*)d_in[19];
    const float *eW=(const float*)d_in[20], *eb_=(const float*)d_in[21];
    const float *rW=(const float*)d_in[22], *rb=(const float*)d_in[23];

    char* base = (char*)d_ws;
    float*  Arow   = (float*) carveB(base, (size_t)N * 256 * 4);  // aliased: h2g
    float*  Acol   = (float*) carveB(base, (size_t)N * 256 * 4);
    float*  accI   = (float*) carveB(base, (size_t)N * 256 * 4);
    float*  accO   = (float*) carveB(base, (size_t)N * 256 * 4);
    float*  Pe     = (float*) carveB(base, (size_t)N * 128 * 4);
    float*  Gtab   = (float*) carveB(base, (size_t)(R + 1) * 256 * 4);
    float*  evals  = (float*) carveB(base, (size_t)E * 2 * 4);
    float*  denom  = (float*) carveB(base, (size_t)N * 2 * 4);
    __bf16* Hbf    = (__bf16*)carveB(base, (size_t)N * 256 * 2);  // GEMM source
    __bf16* Wbf    = (__bf16*)carveB(base, (size_t)256 * 640 * 2);
    __bf16* gauxbf = (__bf16*)carveB(base, (size_t)(R + 1) * 128 * 2);
    float*  h2g    = Arow;    // dead by the time the level-2 gate writes it

    int nodeBlocks = (N + 7) / 8;

    // x <- l2norm(x) straight to bf16; relation aux table (bf16) with zero row R
    l2norm_rows_bf<<<nodeBlocks, 256, 0, stream>>>(x, Hbf, N);
    build_gaux_bf<<<((R + 1) * 128 + 255) / 256, 256, 0, stream>>>(g, gauxbf, R);

    // Entity residual projection Pe = xn@eW.T + eb ; relation output g@rW.T + rb
    conv_bf16(eW, Wbf, 128L * 128, stream);
    launch_gemm<128, 128, true>(Hbf, N, Wbf, 128, eb_, Pe, stream);
    conv_bf16(rW, Wbf, 128L * 128, stream);
    launch_gemm<128, 128, true>(gauxbf, R, Wbf, 128, rb,
                                (float*)d_out + (size_t)N * 128, stream);

    // ---- level 1 ----
    att_dir(Hbf, 128, W_ii, b_ii, att_ii, erow, ecol, ek1, ek2,
            gauxbf, Wbf, Arow, Acol, Gtab, evals, denom, accI, N, R, E, stream);
    att_dir(Hbf, 128, W_io, b_io, att_io, ecol, erow, ek1, ek2,
            gauxbf, Wbf, Arow, Acol, Gtab, evals, denom, accO, N, R, E, stream);
    gate_norm<true><<<nodeBlocks, 256, 0, stream>>>(accI, accO, aW_in, ab_in,
                                                    (void*)Hbf, N);

    // ---- level 2 (source features now bf16 [N,256] in Hbf) ----
    att_dir(Hbf, 256, W_oi, b_oi, att_oi, erow, ecol, ek1, ek2,
            gauxbf, Wbf, Arow, Acol, Gtab, evals, denom, accI, N, R, E, stream);
    att_dir(Hbf, 256, W_oo, b_oo, att_oo, ecol, erow, ek1, ek2,
            gauxbf, Wbf, Arow, Acol, Gtab, evals, denom, accO, N, R, E, stream);
    gate_norm<false><<<nodeBlocks, 256, 0, stream>>>(accI, accO, aW_out, ab_out,
                                                     (void*)h2g, N);

    // ---- final merge ----
    final_merge<<<nodeBlocks, 256, 0, stream>>>(Pe, h2g, (float*)d_out, N);
}
